// NodeTaskHead_FitSphere_35107062677960
// MI455X (gfx1250) — compile-verified, matrix operand order checked by reading, not run on hardware
//
#include <hip/hip_runtime.h>
#include <hip/hip_bf16.h>
#include <math.h>

// ---------------- CDNA5 WMMA types ----------------
typedef __attribute__((ext_vector_type(16))) __bf16 v16bf;
typedef __attribute__((ext_vector_type(8)))  float  v8f;

union FragBF { v16bf v; unsigned int u[8]; };

__device__ __forceinline__ unsigned short f2bf(float f) {
  unsigned int x = __float_as_uint(f);
  x += 0x7FFFu + ((x >> 16) & 1u);          // round-to-nearest-even
  return (unsigned short)(x >> 16);
}

// Pack two f32 into a bf16x2 dword.
// Preference: HW v_cvt_pk_bf16_f32 > v_perm_b32 round-half-up (1 add/elem + 1 perm/pair)
// > manual RNE.  Half-up differs from RNE only on exact 16-bit ties.
__device__ __forceinline__ unsigned int pack_bf2(float a, float b) {
#if __has_builtin(__builtin_amdgcn_cvt_pk_bf16_f32)
  auto t = __builtin_amdgcn_cvt_pk_bf16_f32(a, b);
  return __builtin_bit_cast(unsigned int, t);
#elif __has_builtin(__builtin_amdgcn_perm)
  unsigned int ua = __float_as_uint(a) + 0x8000u;
  unsigned int ub = __float_as_uint(b) + 0x8000u;
  // result = { ub[31:16], ua[31:16] } : sel bytes 7,6 from S0=ub ; 3,2 from S1=ua
  return __builtin_amdgcn_perm(ub, ua, 0x07060302u);
#else
  return (unsigned int)f2bf(a) | ((unsigned int)f2bf(b) << 16);
#endif
}

__device__ __forceinline__ float gelu_exact(float x) {
  return 0.5f * x * (1.0f + erff(x * 0.70710678118654752f));
}

// Load a 16x32 f32 tile (row-major, row stride ld) as a bf16 WMMA fragment.
// ISA 7.12.2 16-bit A-matrix 16x32: lane r=lane&15 holds row r, half h=lane>>4;
// VGPR0..3 = K = h*8 + {0..7} (2/VGPR); VGPR4..7 = K = 16 + h*8 + {0..7}.
// B (32x16) uses the identical per-lane packing with "row" = column N = row of W.
template <bool NT>
__device__ __forceinline__ v16bf load_frag_f32(const float* __restrict__ base, int ld, int lane) {
  int r = lane & 15, h = lane >> 4;
  const float* p = base + r * ld + h * 8;
  FragBF f;
#pragma unroll
  for (int i = 0; i < 4; ++i) {
    float a0, a1, b0, b1;
    if (NT) {
      a0 = __builtin_nontemporal_load(p + 2 * i);
      a1 = __builtin_nontemporal_load(p + 2 * i + 1);
      b0 = __builtin_nontemporal_load(p + 16 + 2 * i);
      b1 = __builtin_nontemporal_load(p + 16 + 2 * i + 1);
    } else {
      a0 = p[2 * i];      a1 = p[2 * i + 1];
      b0 = p[16 + 2 * i]; b1 = p[16 + 2 * i + 1];
    }
    f.u[i]     = pack_bf2(a0, a1);
    f.u[4 + i] = pack_bf2(b0, b1);
  }
  return f.v;
}

// ---------------- Generic small GEMM: C = act(A @ W^T + bias) ----------------
// A:[M,K] lda, W:[Nout,K] ldw.  TRC=0: C[m*ldc+n].  TRC=1: C[n*ldc+m] (transposed store).
// grid=(Nout/16, M/16), block=32 (1 wave).
template <int ACT, int TRC>
__global__ void gemm_bias_act(const float* __restrict__ A, const float* __restrict__ W,
                              const float* __restrict__ bias, float* __restrict__ C,
                              int K, int lda, int ldw, int ldc) {
  const int n0 = blockIdx.x * 16;
  const int m0 = blockIdx.y * 16;
  const int lane = threadIdx.x & 31;
  v8f acc = {};
  for (int k0 = 0; k0 < K; k0 += 32) {
    v16bf af = load_frag_f32<false>(A + m0 * lda + k0, lda, lane);
    v16bf bf = load_frag_f32<false>(W + n0 * ldw + k0, ldw, lane);
    acc = __builtin_amdgcn_wmma_f32_16x16x32_bf16(false, af, false, bf,
                                                  (short)0, acc, false, false);
  }
  const int n = n0 + (lane & 15);
  const int h = lane >> 4;
  const float bv = bias ? bias[n] : 0.0f;
#pragma unroll
  for (int v = 0; v < 8; ++v) {
    int m = m0 + v + 8 * h;                    // C/D layout: VGPR v -> M=v (+8 for upper half)
    float x = acc[v] + bv;
    if (ACT) x = gelu_exact(x);
    if (TRC) C[n * ldc + m] = x;
    else     C[m * ldc + n] = x;
  }
}

// ---------------- Fused edge kernel ----------------
// One workgroup per (b,i).  8 waves; wave w handles j-tiles w*4..w*4+3 (16 j each).
// t[j,c] = (gbf[b,i,j,:] @ Wc^T)[c] + Aq[b,i,c] + BkT[c, b*512+j]  (Aq includes e_b1)
// u = gelu(t); fsp = u . e_w2 + e_b2; masks; fsp_d out; moment sums -> 3x3 solve.
__global__ void __launch_bounds__(256)
edge_kernel(const float* __restrict__ gbf, const float* __restrict__ dpos,
            const int* __restrict__ etm, const unsigned char* __restrict__ dem,
            const float* __restrict__ e_w1, const float* __restrict__ e_w2,
            const float* __restrict__ e_b2,
            const float* __restrict__ Aq, const float* __restrict__ BkT,
            float* __restrict__ out_pos, float* __restrict__ out_fsp) {
  __shared__ unsigned int sW[128 * 65];     // Wc as bf16 pairs, padded rows (65 dwords)
  __shared__ float sStats[11][128];
  __shared__ float sAcc[11];

  const int tid  = threadIdx.x;
  const int bi   = blockIdx.x;              // b*512 + i
  const int b    = bi >> 9;
  const int i    = bi & 511;

  // Stage Wc = e_w1[:, 128:256] into LDS as bf16.
  for (int p = tid; p < 8192; p += 256) {   // 128 rows * 64 dword-pairs
    int c = p >> 6, gp = p & 63;
    const float* src = e_w1 + c * 256 + 128 + 2 * gp;
    sW[c * 65 + gp] = pack_bf2(src[0], src[1]);
  }
  __syncthreads();

  const int wave = tid >> 5, lane = tid & 31;
  const int r = lane & 15, h = lane >> 4;
  const float b2v = e_b2[0];
  const bool activeLane = (r < 8);

  float w2c[8], aqc[8];
#pragma unroll
  for (int ns = 0; ns < 8; ++ns) {
    int c = ns * 16 + r;
    w2c[ns] = e_w2[c];
    aqc[ns] = Aq[bi * 128 + c];
  }

  float cntV = 0, cntK = 0, v3x = 0, v3y = 0, v3z = 0;
  float axx = 0, axy = 0, axz = 0, ayy = 0, ayz = 0, azz = 0;

  for (int it = 0; it < 4; ++it) {
    const int j0 = (wave * 4 + it) * 16;
    const float* gbase = gbf + ((long)bi * 512 + j0) * 128;

    v16bf a[4];
#pragma unroll
    for (int ks = 0; ks < 4; ++ks) a[ks] = load_frag_f32<true>(gbase + ks * 32, 128, lane);

    v8f acc[8];
#pragma unroll
    for (int ns = 0; ns < 8; ++ns) acc[ns] = (v8f){};
#pragma unroll
    for (int ns = 0; ns < 8; ++ns) {
#pragma unroll
      for (int ks = 0; ks < 4; ++ks) {
        FragBF bf;
        int baseu = (ns * 16 + r) * 65 + ks * 16 + h * 4;   // (ks*32 + h*8)/2
#pragma unroll
        for (int q = 0; q < 4; ++q) { bf.u[q] = sW[baseu + q]; bf.u[4 + q] = sW[baseu + 8 + q]; }
        acc[ns] = __builtin_amdgcn_wmma_f32_16x16x32_bf16(false, a[ks], false, bf.v,
                                                          (short)0, acc[ns], false, false);
      }
    }

    // Epilogue: gelu + dot with e_w2 over this lane's 8 channels, then butterfly.
    // BkT is [128 c][1024 node]: this lane's 8 j values are contiguous -> 2x float4.
    const int nodeBase = b * 512 + j0 + 8 * h;   // + v, v=0..7
    float partial[8];
#pragma unroll
    for (int v = 0; v < 8; ++v) partial[v] = 0.0f;
#pragma unroll
    for (int ns = 0; ns < 8; ++ns) {
      int c = ns * 16 + r;
      const float4* bkp = (const float4*)(BkT + (long)c * 1024 + nodeBase);
      float4 bk0 = bkp[0];
      float4 bk1 = bkp[1];
      float bkv[8] = {bk0.x, bk0.y, bk0.z, bk0.w, bk1.x, bk1.y, bk1.z, bk1.w};
#pragma unroll
      for (int v = 0; v < 8; ++v) {
        float t = acc[ns][v] + aqc[ns] + bkv[v];
        partial[v] += gelu_exact(t) * w2c[ns];
      }
    }
#pragma unroll
    for (int v = 0; v < 8; ++v) {           // reduce over 16 lanes within each half
      partial[v] += __shfl_xor(partial[v], 1, 32);
      partial[v] += __shfl_xor(partial[v], 2, 32);
      partial[v] += __shfl_xor(partial[v], 4, 32);
      partial[v] += __shfl_xor(partial[v], 8, 32);
    }

    if (activeLane) {                        // lane r<8 owns j_local = r + 8*h
      float fsp0 = 0.0f;
#pragma unroll
      for (int v = 0; v < 8; ++v) if (v == r) fsp0 = partial[v];
      int   j    = j0 + r + 8 * h;
      long  eidx = (long)bi * 512 + j;
      float fsp  = fsp0 + b2v;
      int   valid = (etm[eidx] != 0);
      float fm   = valid ? fsp : 0.0f;
      int   drop = (dem[i * 512 + j] != 0);
      float fd   = drop ? 0.0f : fm;
      out_fsp[eidx] = fd;
      float px = dpos[eidx * 3 + 0] * fd;
      float py = dpos[eidx * 3 + 1] * fd;
      float pz = dpos[eidx * 3 + 2] * fd;
      float sq = px * px + py * py + pz * pz;
      cntV += (float)valid;
      cntK += (valid && !drop) ? 1.0f : 0.0f;
      v3x += sq * px; v3y += sq * py; v3z += sq * pz;
      axx += px * px; axy += px * py; axz += px * pz;
      ayy += py * py; ayz += py * pz; azz += pz * pz;
    }
  }

  // Deterministic reduction of the 11 per-thread stats (128 active threads).
  if (activeLane) {
    int widx = wave * 16 + h * 8 + r;
    sStats[0][widx] = cntV;  sStats[1][widx] = cntK;
    sStats[2][widx] = v3x;   sStats[3][widx] = v3y;  sStats[4][widx] = v3z;
    sStats[5][widx] = axx;   sStats[6][widx] = axy;  sStats[7][widx] = axz;
    sStats[8][widx] = ayy;   sStats[9][widx] = ayz;  sStats[10][widx] = azz;
  }
  __syncthreads();
  if (tid < 11) {
    float s = 0.0f;
    for (int t = 0; t < 128; ++t) s += sStats[tid][t];
    sAcc[tid] = s;
  }
  __syncthreads();

  if (tid == 0) {
    float cv = sAcc[0], ck = sAcc[1];
    float eff = ck + 1e-5f;
    bool untrust = (eff / (cv + 1e-5f)) < 0.3f;
    float bx = sAcc[2] / eff, by = sAcc[3] / eff, bz = sAcc[4] / eff;
    float m00 = 2.0f * (sAcc[5] / eff + 1e-8f);
    float m01 = 2.0f * (sAcc[6] / eff);
    float m02 = 2.0f * (sAcc[7] / eff);
    float m11 = 2.0f * (sAcc[8] / eff + 1e-8f);
    float m12 = 2.0f * (sAcc[9] / eff);
    float m22 = 2.0f * (sAcc[10] / eff + 1e-8f);
    float c00 = m11 * m22 - m12 * m12;
    float c01 = m02 * m12 - m01 * m22;
    float c02 = m01 * m12 - m02 * m11;
    float c11 = m00 * m22 - m02 * m02;
    float c12 = m01 * m02 - m00 * m12;
    float c22 = m00 * m11 - m01 * m01;
    float det = m00 * c00 + m01 * c01 + m02 * c02;
    float inv = 1.0f / det;
    float x = (c00 * bx + c01 * by + c02 * bz) * inv;
    float y = (c01 * bx + c11 * by + c12 * bz) * inv;
    float z = (c02 * bx + c12 * by + c22 * bz) * inv;
    out_pos[bi * 3 + 0] = untrust ? 0.0f : 2.0f * x;
    out_pos[bi * 3 + 1] = untrust ? 0.0f : 2.0f * y;
    out_pos[bi * 3 + 2] = untrust ? 0.0f : 2.0f * z;
  }
}

// ---------------- Host launcher ----------------
extern "C" void kernel_launch(void* const* d_in, const int* in_sizes, int n_in,
                              void* d_out, int out_size, void* d_ws, size_t ws_size,
                              hipStream_t stream) {
  const float* query = (const float*)d_in[0];
  const float* gbf   = (const float*)d_in[1];
  const float* dpos  = (const float*)d_in[2];
  const int*   etm   = (const int*)d_in[3];
  const unsigned char* dem = (const unsigned char*)d_in[4];
  // d_in[5] = drop_or_add (scalar, always 1 in this branch)
  const float* q_w1 = (const float*)d_in[6];
  const float* q_b1 = (const float*)d_in[7];
  const float* q_w2 = (const float*)d_in[8];
  const float* q_b2 = (const float*)d_in[9];
  const float* k_w1 = (const float*)d_in[10];
  const float* k_b1 = (const float*)d_in[11];
  const float* k_w2 = (const float*)d_in[12];
  const float* k_b2 = (const float*)d_in[13];
  const float* e_w1 = (const float*)d_in[14];
  const float* e_b1 = (const float*)d_in[15];
  const float* e_w2 = (const float*)d_in[16];
  const float* e_b2 = (const float*)d_in[17];

  float* out     = (float*)d_out;
  float* out_pos = out;            // [2,512,3]
  float* out_fsp = out + 3072;     // [2,512,512]

  float* ws  = (float*)d_ws;
  float* hq  = ws;                 // [1024,768]
  float* hk  = hq + 1024 * 768;
  float* gq  = hk + 1024 * 768;    // [1024,64]  gelu(q)
  float* gk  = gq + 1024 * 64;
  float* Aq  = gk + 1024 * 64;     // [1024,128] gelu(q)@Wa^T + e_b1
  float* BkT = Aq + 1024 * 128;    // [128,1024] (gelu(k)@Wb^T)^T

  dim3 blk(32);
  // layer1 of q/k projections (gelu)
  gemm_bias_act<1, 0><<<dim3(48, 64), blk, 0, stream>>>(query, q_w1, q_b1, hq, 768, 768, 768, 768);
  gemm_bias_act<1, 0><<<dim3(48, 64), blk, 0, stream>>>(query, k_w1, k_b1, hk, 768, 768, 768, 768);
  // layer2 fused with the e_emb gelu: gq = gelu(hq@w2^T + b2)
  gemm_bias_act<1, 0><<<dim3(4, 64), blk, 0, stream>>>(hq, q_w2, q_b2, gq, 768, 768, 768, 64);
  gemm_bias_act<1, 0><<<dim3(4, 64), blk, 0, stream>>>(hk, k_w2, k_b2, gk, 768, 768, 768, 64);
  // per-node halves of e_proj layer1 (Wa=e_w1[:, :64], Wb=e_w1[:, 64:128])
  gemm_bias_act<0, 0><<<dim3(8, 64), blk, 0, stream>>>(gq, e_w1,      e_b1,    Aq,  64, 64, 256, 128);
  gemm_bias_act<0, 1><<<dim3(8, 64), blk, 0, stream>>>(gk, e_w1 + 64, nullptr, BkT, 64, 64, 256, 1024);
  // fused per-edge GEMM + gelu + e_w2 dot + masks + sphere-fit
  edge_kernel<<<dim3(1024), dim3(256), 0, stream>>>(gbf, dpos, etm, dem, e_w1, e_w2, e_b2,
                                                    Aq, BkT, out_pos, out_fsp);
}
